// ExpertAttention_39204461478152
// MI455X (gfx1250) — compile-verified, matrix operand order checked by reading, not run on hardware
//
#include <hip/hip_runtime.h>
#include <cstdint>
#include <cstddef>

// Problem constants (BERT-base shaped expert attention)
#define B_   64
#define S_   512
#define D_   768
#define H_   12
#define HD_  64
#define E_   4
#define EF_  3
#define NQKV 2304   // 3*D, fused QKV projection width

typedef __attribute__((ext_vector_type(16))) __bf16 v16bf;
typedef __attribute__((ext_vector_type(8)))  float  v8f;
typedef int v4i_gcc __attribute__((vector_size(16)));   // matches builtin param

#define AS1 __attribute__((address_space(1)))
#define AS3 __attribute__((address_space(3)))

#if __has_builtin(__builtin_amdgcn_global_load_async_to_lds_b128) && \
    __has_builtin(__builtin_amdgcn_s_wait_asynccnt)
#define HAS_ASYNC_LDS 1
#else
#define HAS_ASYNC_LDS 0
#endif

// Async global->LDS 16-byte copy (CDNA5 GLOBAL_LOAD_ASYNC_TO_LDS_B128,
// ASYNCcnt-tracked).  Generic->AS3 via low 32 bits (LDS aperture keeps the
// byte offset in addr[31:0]); generic->AS1 is value-preserving.
__device__ __forceinline__ void async_cp16(const void* g, void* l) {
#if HAS_ASYNC_LDS
    __builtin_amdgcn_global_load_async_to_lds_b128(
        (AS1 v4i_gcc*)(uintptr_t)g,
        (AS3 v4i_gcc*)(unsigned int)(uintptr_t)l, 0, 0);
#else
    *(uint4*)l = *(const uint4*)g;
#endif
}
__device__ __forceinline__ void async_join() {
#if HAS_ASYNC_LDS
    __builtin_amdgcn_s_wait_asynccnt(0);
#endif
}

// WMMA 16x16x32 bf16 fragment: 8 VGPRs of packed bf16 pairs per lane.
union Frag {
    v16bf          bf;
    unsigned short u[16];
    unsigned int   w[8];
};

__device__ __forceinline__ unsigned short f2bf(float f) {
    unsigned u = __float_as_uint(f);
    u += 0x7FFFu + ((u >> 16) & 1u);           // round-to-nearest-even
    return (unsigned short)(u >> 16);
}
__device__ __forceinline__ unsigned short cvt_bf(float v)          { return f2bf(v); }
__device__ __forceinline__ unsigned short cvt_bf(unsigned short v) { return v; }

// K-index of fragment VGPR j (pair base), per the CDNA5 16-bit A/B layout:
// lanes 0-15 cover K={0..7,16..23}, lanes 16-31 cover K={8..15,24..31}.
__device__ __forceinline__ int kform(int j, int hi) {
    return ((j >> 2) << 4) + (hi << 3) + ((j & 3) << 1);
}

__device__ __forceinline__ v8f wmma_bf16(const Frag& a, const Frag& b, v8f c) {
    return __builtin_amdgcn_wmma_f32_16x16x32_bf16(
        false, a.bf, false, b.bf, (short)0, c, false, false);
}

// ---------------------------------------------------------------------------
// Kernel 1: routing.  h = mean_s(hs[b]);  z[b,e] = mean_d |(h-mu)/sigma|;
// assign = first e with z<2 else 3.
// ---------------------------------------------------------------------------
__global__ __launch_bounds__(256) void route_kernel(
        const float* __restrict__ hs, const float* __restrict__ dmean,
        const float* __restrict__ dstd, int* __restrict__ assign) {
    const int b = blockIdx.x, t = threadIdx.x;
    float zpart[EF_] = {0.f, 0.f, 0.f};
    for (int d = t; d < D_; d += 256) {
        const float* p = hs + (size_t)b * S_ * D_ + d;
        float sum = 0.f;
        for (int s = 0; s < S_; ++s) sum += p[(size_t)s * D_];
        const float h = sum * (1.0f / (float)S_);
        #pragma unroll
        for (int e = 0; e < EF_; ++e)
            zpart[e] += fabsf((h - dmean[e * D_ + d]) / dstd[e * D_ + d]);
    }
    __shared__ float red[EF_][256];
    #pragma unroll
    for (int e = 0; e < EF_; ++e) red[e][t] = zpart[e];
    __syncthreads();
    for (int off = 128; off > 0; off >>= 1) {
        if (t < off) {
            #pragma unroll
            for (int e = 0; e < EF_; ++e) red[e][t] += red[e][t + off];
        }
        __syncthreads();
    }
    if (t == 0) {
        int a = E_ - 1;
        for (int e = EF_ - 1; e >= 0; --e)
            if (red[e][0] * (1.0f / (float)D_) < 2.0f) a = e;  // first fit wins
        assign[b] = a;
    }
}

// ---------------------------------------------------------------------------
// Kernels 2-4: one-time (per launch) weight/bias prep: f32 -> bf16,
// Wq|Wk|Wv concatenated along N so QKV is a single GEMM.
// ---------------------------------------------------------------------------
__global__ __launch_bounds__(256) void prep_wqkv(
        const float* __restrict__ Wq, const float* __restrict__ Wk,
        const float* __restrict__ Wv, unsigned short* __restrict__ out) {
    const size_t idx = (size_t)blockIdx.x * 256 + threadIdx.x;
    if (idx >= (size_t)E_ * D_ * NQKV) return;
    const int    n  = (int)(idx % NQKV);
    const size_t ed = idx / NQKV;
    const int    d  = (int)(ed % D_);
    const int    e  = (int)(ed / D_);
    const int which = n / D_, nn = n % D_;
    const float* src = (which == 0) ? Wq : (which == 1) ? Wk : Wv;
    out[idx] = f2bf(src[((size_t)e * D_ + d) * D_ + nn]);
}

__global__ __launch_bounds__(256) void prep_wo(
        const float* __restrict__ Wo, unsigned short* __restrict__ out) {
    const size_t idx = (size_t)blockIdx.x * 256 + threadIdx.x;
    if (idx >= (size_t)E_ * D_ * D_) return;
    out[idx] = f2bf(Wo[idx]);
}

__global__ __launch_bounds__(256) void prep_bias(
        const float* __restrict__ bq, const float* __restrict__ bk,
        const float* __restrict__ bv, float* __restrict__ out) {
    const int idx = blockIdx.x * 256 + threadIdx.x;
    if (idx >= E_ * NQKV) return;
    const int n = idx % NQKV, e = idx / NQKV;
    const int which = n / D_, nn = n % D_;
    const float* src = (which == 0) ? bq : (which == 1) ? bk : bv;
    out[idx] = src[e * D_ + nn];
}

// ---------------------------------------------------------------------------
// Kernel 5/7: expert-gathered GEMM.  C[b] = A[b] (S x 768) @ W[assign[b]]
// (768 x NOUT) + bias[assign[b]].  64x64 block tile, 4 waves x (16 rows, 4
// accumulators).  Double-buffered K slabs: weights streamed with async
// global->LDS (bf16 copy), activations converted f32->bf16 through VGPRs.
// One barrier per K slab.
// ---------------------------------------------------------------------------
template <typename AT, int NOUT, bool OUT_BF16>
__global__ __launch_bounds__(128) void expert_gemm(
        const AT* __restrict__ A, const unsigned short* __restrict__ W,
        const float* __restrict__ bias, void* __restrict__ Out,
        const int* __restrict__ assign) {
    const int b  = blockIdx.z;
    const int s0 = blockIdx.y * 64;
    const int n0 = blockIdx.x * 64;
    const int e  = assign[b];
    const int t  = threadIdx.x;
    const int wave = t >> 5, lane = t & 31, lo = lane & 15, hi = lane >> 4;

    __shared__ unsigned short a_lds[2][64][36];  // 64x32 bf16, ping-pong
    __shared__ unsigned short b_lds[2][32][80];  // 32x64 bf16, 160B rows (16B align)

    const AT*             Ab = A + (size_t)b * S_ * D_;
    const unsigned short* We = W + (size_t)e * D_ * NOUT + n0;

    v8f acc[4];
    #pragma unroll
    for (int i = 0; i < 4; ++i)
        #pragma unroll
        for (int r = 0; r < 8; ++r) acc[i][r] = 0.0f;

    // ---- prologue: stage slab 0 ----
    #pragma unroll
    for (int i = 0; i < 2; ++i) {              // B: 32 rows x 8 chunks of 16B
        const int idx = i * 128 + t, row = idx >> 3, c8 = (idx & 7) * 8;
        async_cp16(We + (size_t)row * NOUT + c8, &b_lds[0][row][c8]);
    }
    #pragma unroll
    for (int i = 0; i < 16; ++i) {             // A: 64x32, convert on the fly
        const int idx = i * 128 + t, row = idx >> 5, col = idx & 31;
        a_lds[0][row][col] = cvt_bf(Ab[(size_t)(s0 + row) * D_ + col]);
    }
    async_join();
    __syncthreads();

    const int NSLAB = D_ / 32;
    for (int ks = 0; ks < NSLAB; ++ks) {
        const int cur = ks & 1, nxt = cur ^ 1;
        const int k1 = (ks + 1) * 32;
        AT areg[16];
        if (ks + 1 < NSLAB) {
            // stream next weight slab straight into LDS (async engine)
            #pragma unroll
            for (int i = 0; i < 2; ++i) {
                const int idx = i * 128 + t, row = idx >> 3, c8 = (idx & 7) * 8;
                async_cp16(We + (size_t)(k1 + row) * NOUT + c8, &b_lds[nxt][row][c8]);
            }
            // issue next activation loads early (hide latency under WMMAs)
            #pragma unroll
            for (int i = 0; i < 16; ++i) {
                const int idx = i * 128 + t, row = idx >> 5, col = idx & 31;
                areg[i] = Ab[(size_t)(s0 + row) * D_ + k1 + col];
            }
        }

        // ---- compute current slab ----
        Frag af;
        const int m = wave * 16 + lo;
        #pragma unroll
        for (int j = 0; j < 8; ++j)
            af.w[j] = *(const unsigned int*)&a_lds[cur][m][kform(j, hi)];
        #pragma unroll
        for (int nt = 0; nt < 4; ++nt) {
            Frag bf_;
            const int n = nt * 16 + lo;
            #pragma unroll
            for (int j = 0; j < 8; ++j) {
                const int k = kform(j, hi);
                bf_.u[2 * j]     = b_lds[cur][k][n];
                bf_.u[2 * j + 1] = b_lds[cur][k + 1][n];
            }
            acc[nt] = wmma_bf16(af, bf_, acc[nt]);
        }

        if (ks + 1 < NSLAB) {
            #pragma unroll
            for (int i = 0; i < 16; ++i) {
                const int idx = i * 128 + t, row = idx >> 5, col = idx & 31;
                a_lds[nxt][row][col] = cvt_bf(areg[i]);
            }
        }
        async_join();
        __syncthreads();
    }

    #pragma unroll
    for (int nt = 0; nt < 4; ++nt) {
        #pragma unroll
        for (int r = 0; r < 8; ++r) {
            const int grow = s0 + wave * 16 + r + (hi << 3);
            const int gcol = n0 + nt * 16 + lo;
            const float v  = acc[nt][r] + bias[(size_t)e * NOUT + gcol];
            const size_t oidx = ((size_t)b * S_ + grow) * NOUT + gcol;
            if constexpr (OUT_BF16) ((unsigned short*)Out)[oidx] = f2bf(v);
            else                    ((float*)Out)[oidx] = v;
        }
    }
}

// ---------------------------------------------------------------------------
// Kernel 6: fused flash-style attention over one (b, h, 64-query) tile.
// Q frags pinned in registers; loop over 64-key tiles with online softmax.
// K/V tiles double-buffered via async global->LDS; score tensor never
// touches HBM (saves ~1.6 GB of traffic vs. materializing it).
// ---------------------------------------------------------------------------
__global__ __launch_bounds__(128) void expert_attn(
        const unsigned short* __restrict__ QKV, const float* __restrict__ amask,
        unsigned short* __restrict__ Ctx) {
    const int b = blockIdx.z, h = blockIdx.y, q0 = blockIdx.x * 64;
    const int t = threadIdx.x;
    const int wave = t >> 5, lane = t & 31, lo = lane & 15, hi = lane >> 4;

    __shared__ unsigned short q_lds[64][80];
    __shared__ unsigned short k_lds[2][64][80];
    __shared__ unsigned short v_lds[2][64][80];
    __shared__ unsigned short p_lds[64][80];
    __shared__ float mask_lds[S_];

    const unsigned short* Qb = QKV + (size_t)b * S_ * NQKV + h * HD_;
    const unsigned short* Kb = Qb + D_;
    const unsigned short* Vb = Qb + 2 * D_;

    // Whole additive key-mask row once (2 KB LDS).
    #pragma unroll
    for (int i = 0; i < 4; ++i)
        mask_lds[i * 128 + t] = amask[b * S_ + i * 128 + t];

    // Stage Q tile (64 queries x 64 head-dim) and key tile 0 / value tile 0.
    #pragma unroll
    for (int i = 0; i < 4; ++i) {
        const int idx = i * 128 + t, row = idx >> 3, c8 = (idx & 7) * 8;
        async_cp16(Kb + (size_t)row * NQKV + c8, &k_lds[0][row][c8]);
        async_cp16(Vb + (size_t)row * NQKV + c8, &v_lds[0][row][c8]);
    }
    #pragma unroll
    for (int i = 0; i < 16; ++i) {
        const int idx = i * 128 + t, row = idx >> 5, cu = idx & 31;
        *(unsigned int*)&q_lds[row][cu * 2] =
            *(const unsigned int*)&Qb[(size_t)(q0 + row) * NQKV + cu * 2];
    }
    async_join();
    __syncthreads();

    Frag qa[2];
    {
        const int m = wave * 16 + lo;
        #pragma unroll
        for (int ks = 0; ks < 2; ++ks)
            #pragma unroll
            for (int j = 0; j < 8; ++j)
                qa[ks].w[j] = *(const unsigned int*)&q_lds[m][ks * 32 + kform(j, hi)];
    }

    float mrow[8], lrow[8];
    #pragma unroll
    for (int r = 0; r < 8; ++r) { mrow[r] = -1e30f; lrow[r] = 0.f; }
    v8f o[4];
    #pragma unroll
    for (int i = 0; i < 4; ++i)
        #pragma unroll
        for (int r = 0; r < 8; ++r) o[i][r] = 0.0f;

    for (int jt = 0; jt < S_ / 64; ++jt) {
        const int cur = jt & 1, j0 = jt * 64;
        if (jt + 1 < S_ / 64) {   // stream next K/V tiles under this tile's math
            const int j1 = j0 + 64;
            #pragma unroll
            for (int i = 0; i < 4; ++i) {
                const int idx = i * 128 + t, row = idx >> 3, c8 = (idx & 7) * 8;
                async_cp16(Kb + (size_t)(j1 + row) * NQKV + c8, &k_lds[cur ^ 1][row][c8]);
                async_cp16(Vb + (size_t)(j1 + row) * NQKV + c8, &v_lds[cur ^ 1][row][c8]);
            }
        }

        // S = Q @ K^T : B-frag element (k=hd, n=key) = K[key][hd]
        v8f sc[4];
        #pragma unroll
        for (int i = 0; i < 4; ++i)
            #pragma unroll
            for (int r = 0; r < 8; ++r) sc[i][r] = 0.0f;
        #pragma unroll
        for (int ks = 0; ks < 2; ++ks)
            #pragma unroll
            for (int nt = 0; nt < 4; ++nt) {
                Frag bf_;
                const int n = nt * 16 + lo;  // key column
                #pragma unroll
                for (int j = 0; j < 8; ++j)
                    bf_.w[j] = *(const unsigned int*)&k_lds[cur][n][ks * 32 + kform(j, hi)];
                sc[nt] = wmma_bf16(qa[ks], bf_, sc[nt]);
            }

        // scale + additive key mask + online softmax (rows live in 16-lane
        // halves; shfl_xor masks < 16 stay within a half).
        float rmax[8];
        #pragma unroll
        for (int r = 0; r < 8; ++r) rmax[r] = -1e30f;
        #pragma unroll
        for (int nt = 0; nt < 4; ++nt) {
            const float mk = mask_lds[j0 + nt * 16 + lo];
            #pragma unroll
            for (int r = 0; r < 8; ++r) {
                const float s = sc[nt][r] * 0.125f + mk;  // 1/sqrt(64)
                sc[nt][r] = s;
                rmax[r] = fmaxf(rmax[r], s);
            }
        }
        #pragma unroll
        for (int r = 0; r < 8; ++r) {
            #pragma unroll
            for (int msk = 8; msk > 0; msk >>= 1)
                rmax[r] = fmaxf(rmax[r], __shfl_xor(rmax[r], msk, 32));
        }
        float scl[8], rsum[8];
        #pragma unroll
        for (int r = 0; r < 8; ++r) {
            const float mn = fmaxf(mrow[r], rmax[r]);
            scl[r]  = __expf(mrow[r] - mn);
            mrow[r] = mn;
            rsum[r] = 0.f;
        }
        #pragma unroll
        for (int nt = 0; nt < 4; ++nt)
            #pragma unroll
            for (int r = 0; r < 8; ++r) {
                const float p = __expf(sc[nt][r] - mrow[r]);
                rsum[r] += p;
                p_lds[wave * 16 + r + (hi << 3)][nt * 16 + lo] = f2bf(p);
                o[nt][r] *= scl[r];
            }
        #pragma unroll
        for (int r = 0; r < 8; ++r) {
            #pragma unroll
            for (int msk = 8; msk > 0; msk >>= 1)
                rsum[r] += __shfl_xor(rsum[r], msk, 32);
            lrow[r] = lrow[r] * scl[r] + rsum[r];
        }

        // O += P @ V  (p_lds rows are wave-local; no cross-wave barrier needed)
        #pragma unroll
        for (int ks = 0; ks < 2; ++ks) {
            Frag pa;
            const int m = wave * 16 + lo;
            #pragma unroll
            for (int j = 0; j < 8; ++j)
                pa.w[j] = *(const unsigned int*)&p_lds[m][ks * 32 + kform(j, hi)];
            #pragma unroll
            for (int nt = 0; nt < 4; ++nt) {
                Frag vb;
                const int n = nt * 16 + lo;  // head-dim column
                #pragma unroll
                for (int j = 0; j < 8; ++j) {
                    const int k = ks * 32 + kform(j, hi);
                    vb.u[2 * j]     = v_lds[cur][k][n];
                    vb.u[2 * j + 1] = v_lds[cur][k + 1][n];
                }
                o[nt] = wmma_bf16(pa, vb, o[nt]);
            }
        }
        async_join();
        __syncthreads();
    }

    #pragma unroll
    for (int nt = 0; nt < 4; ++nt)
        #pragma unroll
        for (int r = 0; r < 8; ++r) {
            const float v  = o[nt][r] / lrow[r];
            const int grow = q0 + wave * 16 + r + (hi << 3);
            const int gcol = h * HD_ + nt * 16 + lo;
            Ctx[((size_t)b * S_ + grow) * D_ + gcol] = f2bf(v);
        }
}

// ---------------------------------------------------------------------------
extern "C" void kernel_launch(void* const* d_in, const int* in_sizes, int n_in,
                              void* d_out, int out_size, void* d_ws, size_t ws_size,
                              hipStream_t stream) {
    const float* hs = (const float*)d_in[0];
    const float* am = (const float*)d_in[1];
    const float* Wq = (const float*)d_in[2];
    const float* bq = (const float*)d_in[3];
    const float* Wk = (const float*)d_in[4];
    const float* bk = (const float*)d_in[5];
    const float* Wv = (const float*)d_in[6];
    const float* bv = (const float*)d_in[7];
    const float* Wo = (const float*)d_in[8];
    const float* bo = (const float*)d_in[9];
    const float* dm = (const float*)d_in[10];
    const float* ds = (const float*)d_in[11];

    char* ws = (char*)d_ws;
    size_t off = 0;
    auto alloc = [&](size_t bytes) -> void* {
        void* p = ws + off;
        off = (off + bytes + 255) & ~(size_t)255;
        return p;
    };
    int*            assign = (int*)alloc((size_t)B_ * sizeof(int));
    unsigned short* wqkv   = (unsigned short*)alloc((size_t)E_ * D_ * NQKV * 2);
    unsigned short* wo     = (unsigned short*)alloc((size_t)E_ * D_ * D_ * 2);
    float*          bqkv   = (float*)alloc((size_t)E_ * NQKV * sizeof(float));
    unsigned short* qkv    = (unsigned short*)alloc((size_t)B_ * S_ * NQKV * 2);
    unsigned short* ctx    = (unsigned short*)alloc((size_t)B_ * S_ * D_ * 2);

    route_kernel<<<B_, 256, 0, stream>>>(hs, dm, ds, assign);
    prep_wqkv<<<((size_t)E_ * D_ * NQKV + 255) / 256, 256, 0, stream>>>(Wq, Wk, Wv, wqkv);
    prep_wo<<<((size_t)E_ * D_ * D_ + 255) / 256, 256, 0, stream>>>(Wo, wo);
    prep_bias<<<(E_ * NQKV + 255) / 256, 256, 0, stream>>>(bq, bk, bv, bqkv);

    expert_gemm<float, NQKV, true>
        <<<dim3(NQKV / 64, S_ / 64, B_), 128, 0, stream>>>(hs, wqkv, bqkv, (void*)qkv, assign);

    expert_attn<<<dim3(S_ / 64, H_, B_), 128, 0, stream>>>(qkv, am, ctx);

    expert_gemm<unsigned short, D_, false>
        <<<dim3(D_ / 64, S_ / 64, B_), 128, 0, stream>>>(ctx, wo, bo, d_out, assign);
}
